// HyenaFilter2d_76355928588834
// MI455X (gfx1250) — matrix operand order repaired and checked by reading, not compiled, not run on hardware
//
#include <hip/hip_runtime.h>
#include <hip/hip_bf16.h>

typedef __bf16 bf16_t;
typedef __attribute__((ext_vector_type(16))) __bf16 v16bf;
typedef __attribute__((ext_vector_type(8)))  __bf16 v8bf;
typedef __attribute__((ext_vector_type(8)))  float  v8f;

namespace {
constexpr int   LL   = 256;   // SEQ
constexpr int   DM   = 64;    // d_model
constexpr int   NB   = 4;     // batch
constexpr int   EMBN = 33;
constexpr int   ORD  = 64;
constexpr float MIN_DEC = -3.0701134573253944f;   // log(0.01)/1.5
constexpr float MAX_DEC = -15.350567286626972f;   // log(0.01)/0.3
}

// Async Global->LDS copy of 16B per lane (tracked by ASYNCcnt, no VGPR data).
// Generic pointers to LDS carry the LDS byte offset in addr[31:0].
__device__ __forceinline__ void async_copy_b128(const uint4* gsrc, uint4* ldst) {
    asm volatile("global_load_async_to_lds_b128 %0, %1, off"
                 :: "v"((unsigned)(unsigned long long)ldst), "v"(gsrc)
                 : "memory");
}
__device__ __forceinline__ void wait_async0() {
    asm volatile("s_wait_asynccnt 0" ::: "memory");
}

// ---------------------------------------------------------------------------
// Kernel 1: implicit filter MLP + exponential decay.
// out[d*L + l] (bf16).  Block = 4 positions, thread = (pos, order) pair.
// ---------------------------------------------------------------------------
__global__ __launch_bounds__(256) void hyena_filters(
    const float* __restrict__ z,
    const float* __restrict__ w1, const float* __restrict__ b1,
    const float* __restrict__ w2, const float* __restrict__ b2,
    const float* __restrict__ w3, const float* __restrict__ b3,
    const float* __restrict__ wo, const float* __restrict__ freq,
    bf16_t* __restrict__ hout, float scale)
{
    __shared__ float sA[4][ORD];
    __shared__ float sB[4][ORD];

    const int o  = threadIdx.x & 63;
    const int lq = threadIdx.x >> 6;             // 0..3
    const int l  = (blockIdx.x & 63) * 4 + lq;   // position 0..255
    const float fo = freq[o];

    float s = b1[o];
    #pragma unroll
    for (int e = 0; e < EMBN; ++e) s += z[l * EMBN + e] * w1[e * ORD + o];
    sA[lq][o] = sinf(fo * s);
    __syncthreads();

    s = b2[o];
    #pragma unroll 8
    for (int k = 0; k < ORD; ++k) s += sA[lq][k] * w2[k * ORD + o];
    sB[lq][o] = sinf(fo * s);
    __syncthreads();

    s = b3[o];
    #pragma unroll 8
    for (int k = 0; k < ORD; ++k) s += sB[lq][k] * w3[k * ORD + o];
    sA[lq][o] = sinf(fo * s);
    __syncthreads();

    s = 0.0f;
    #pragma unroll 8
    for (int k = 0; k < ORD; ++k) s += sA[lq][k] * wo[k * DM + o];

    const float t     = (float)l * (1.0f / (float)(LL - 1));
    const float delta = MIN_DEC + (MAX_DEC - MIN_DEC) * ((float)o / (float)(DM - 1));
    hout[o * LL + l] = (bf16_t)(s * __expf(-t * fabsf(delta)) * scale);
}

// ---------------------------------------------------------------------------
// Kernel 2: stage 1 — causal conv along the LAST axis with hy (per channel).
// A = x tile (bf16 from b128 loads); B = Toeplitz(hy).  All B fragments for a
// block depend only on diag = n - 2*kc (16 distinct) -> prebuilt LDS table in
// WMMA per-lane layout: one aligned v16bf load per site.
// Y1 written bf16 in fragment layout [g][n][m][e] into the plane's own 256KB
// slot of d_out (first half of it).
// ---------------------------------------------------------------------------
__global__ __launch_bounds__(256) void conv_cols(
    const float* __restrict__ x, const bf16_t* __restrict__ hy,
    float* __restrict__ out)
{
    __shared__ bf16_t sh_pad[2 * LL];            // 256 zeros + hy[d]
    __shared__ bf16_t sh_TB[16 * 32 * 16];       // [diag][lane][elem], 16 KB

    const int bd   = blockIdx.x >> 1;            // b*64 + d
    const int half = blockIdx.x & 1;
    const int d    = bd & (DM - 1);
    const int tid  = threadIdx.x;
    const int wave = tid >> 5;
    const int lane = tid & 31;
    const int m    = lane & 15;
    const int hl   = lane >> 4;
    const int g    = half * 8 + wave;            // row-tile 0..15
    const int i0   = g * 16;

    sh_pad[tid]      = (bf16_t)0.0f;
    sh_pad[LL + tid] = hy[d * LL + tid];
    __syncthreads();

    // Build Toeplitz B table: slice s = (diag, lane); elem e -> K = e + 16*hl'
    #pragma unroll
    for (int q = 0; q < 2; ++q) {
        const int s   = tid + q * 256;           // 0..511
        const int dgi = s >> 5;
        const int ln  = s & 31;
        const int mp  = ln & 15;
        const int hp  = ln >> 4;
        const int base = LL + 16 * dgi + mp - 16 * hp;
        #pragma unroll
        for (int e = 0; e < 16; ++e)
            sh_TB[s * 16 + e] = sh_pad[base - e];
    }
    __syncthreads();

    const float* xrow = x + ((size_t)bd * LL + (i0 + m)) * LL;

    v8f acc[16];
    #pragma unroll
    for (int n = 0; n < 16; ++n) { v8f zz = {}; acc[n] = zz; }

    #pragma unroll
    for (int kc = 0; kc < 8; ++kc) {
        // A fragment: A[m][K], K = e + 16*hl
        const float4* p4 = (const float4*)(xrow + kc * 32 + hl * 16);
        const float4 f0 = p4[0], f1 = p4[1], f2 = p4[2], f3 = p4[3];
        v16bf A;
        A[0]  = (bf16_t)f0.x; A[1]  = (bf16_t)f0.y; A[2]  = (bf16_t)f0.z; A[3]  = (bf16_t)f0.w;
        A[4]  = (bf16_t)f1.x; A[5]  = (bf16_t)f1.y; A[6]  = (bf16_t)f1.z; A[7]  = (bf16_t)f1.w;
        A[8]  = (bf16_t)f2.x; A[9]  = (bf16_t)f2.y; A[10] = (bf16_t)f2.z; A[11] = (bf16_t)f2.w;
        A[12] = (bf16_t)f3.x; A[13] = (bf16_t)f3.y; A[14] = (bf16_t)f3.z; A[15] = (bf16_t)f3.w;

        #pragma unroll
        for (int dg = 0; dg < 16; ++dg) {
            if (2 * kc + dg < 16) {              // compile-time causal guard
                const int n = 2 * kc + dg;
                const v16bf Bf = *(const v16bf*)&sh_TB[(dg * 32 + lane) * 16];
                acc[n] = __builtin_amdgcn_wmma_f32_16x16x32_bf16(
                    false, A, false, Bf, (short)0, acc[n], false, false);
            }
        }
    }

    // Epilogue: bf16 fragment layout, 16B contiguous per (tile, lane)
    bf16_t* yf = (bf16_t*)(out + (size_t)bd * LL * LL);   // first 128KB of slot
    #pragma unroll
    for (int n = 0; n < 16; ++n) {
        v8bf pk;
        #pragma unroll
        for (int r = 0; r < 8; ++r) pk[r] = (bf16_t)acc[n][r];
        *(v8bf*)(yf + ((((g * 16 + n) * 16 + m) << 4) + 8 * hl)) = pk;
    }
}

// ---------------------------------------------------------------------------
// Kernel 3: stage 2 — causal conv along the FIRST axis with hx, in place over
// d_out, fused epilogue  out = acc + x*bias[d].
// One 512-thread block per (b,d) plane.  A = Toeplitz(hx) from prebuilt LDS
// table (diag = i_tile - 2*kc).  Y1 chunks (fragment layout) are brought in
// with double-buffered async Global->LDS copies (ASYNCcnt) overlapped with
// the WMMAs of the previous chunk.
// ---------------------------------------------------------------------------
__global__ __launch_bounds__(512) void conv_rows(
    float* __restrict__ out,                 // plane slot: read Y1f / write f32
    const bf16_t* __restrict__ hx,
    const float* __restrict__ x, const float* __restrict__ bias)
{
    __shared__ bf16_t sh_pad[2 * LL];        // 256 zeros + hx[d]
    __shared__ bf16_t sh_TA[16 * 32 * 16];   // A-fragment table, 16 KB
    __shared__ bf16_t sh_ch[2][2 * 16 * 16 * 16]; // double-buffered chunks, 2x16KB

    const int bd   = blockIdx.x;             // b*64 + d
    const int d    = bd & (DM - 1);
    const int tid  = threadIdx.x;
    const int wave = tid >> 5;               // i_tile 0..15
    const int lane = tid & 31;
    const int m    = lane & 15;
    const int hl   = lane >> 4;
    const int i0   = wave * 16;

    if (tid < LL) sh_pad[tid] = (bf16_t)0.0f;
    else          sh_pad[tid] = hx[d * LL + (tid - LL)];
    __syncthreads();

    // Build Toeplitz A table: slot s -> K = s + 16*hl'
    {
        const int sl  = tid;                 // 512 slices, one per thread
        const int dgi = sl >> 5;
        const int ln  = sl & 31;
        const int mp  = ln & 15;
        const int hp  = ln >> 4;
        const int base = LL + 16 * dgi + mp - 16 * hp;
        #pragma unroll
        for (int e = 0; e < 16; ++e) {
            sh_TA[sl * 16 + e]     = sh_pad[base - e];
        }
    }

    bf16_t* yf = (bf16_t*)(out + (size_t)bd * LL * LL);

    v8f acc[16];
    #pragma unroll
    for (int n = 0; n < 16; ++n) { v8f zz = {}; acc[n] = zz; }

    // Prologue: async-load chunk 0 (g-slabs 0,1) into buffer 0
    {
        const uint4* src = (const uint4*)yf;
        uint4* dst = (uint4*)sh_ch[0];
        async_copy_b128(src + tid,       dst + tid);
        async_copy_b128(src + tid + 512, dst + tid + 512);
        wait_async0();
    }
    __syncthreads();                          // chunk0 + tables visible

    #pragma unroll
    for (int kc = 0; kc < 8; ++kc) {
        // Prefetch next chunk into the other buffer (its readers finished at
        // the trailing barrier of iteration kc-1, before this issue).
        if (kc < 7) {
            const uint4* src = (const uint4*)(yf + (size_t)(kc + 1) * 8192);
            uint4* dst = (uint4*)sh_ch[(kc + 1) & 1];
            async_copy_b128(src + tid,       dst + tid);
            async_copy_b128(src + tid + 512, dst + tid + 512);
        }

        if (2 * kc <= wave) {                // causal: A tile non-zero
            const int dg = wave - 2 * kc;    // wave-uniform diag
            const v16bf A = *(const v16bf*)&sh_TA[(dg * 32 + lane) * 16];
            const bf16_t* ch = sh_ch[kc & 1];
            #pragma unroll
            for (int n = 0; n < 16; ++n) {
                const v16bf Bf =
                    *(const v16bf*)&ch[(((hl * 16 + n) * 16 + m) << 4)];
                acc[n] = __builtin_amdgcn_wmma_f32_16x16x32_bf16(
                    false, A, false, Bf, (short)0, acc[n], false, false);
            }
        }

        wait_async0();                        // next chunk landed in LDS
        __syncthreads();                      // all reads of current chunk done
    }

    // Epilogue (all plane reads complete before last barrier -> in-place safe)
    const float bs = bias[d];
    const float* xb = x + (size_t)bd * LL * LL;
    float* ob = out + (size_t)bd * LL * LL;
    #pragma unroll
    for (int n = 0; n < 16; ++n) {
        #pragma unroll
        for (int r = 0; r < 8; ++r) {
            const int row = i0 + r + 8 * hl;
            const int col = 16 * n + m;
            const size_t idx = (size_t)row * LL + col;
            ob[idx] = acc[n][r] + xb[idx] * bs;
        }
    }
}

// ---------------------------------------------------------------------------
extern "C" void kernel_launch(void* const* d_in, const int* in_sizes, int n_in,
                              void* d_out, int out_size, void* d_ws, size_t ws_size,
                              hipStream_t stream) {
    const float* x     = (const float*)d_in[0];
    const float* z     = (const float*)d_in[1];
    const float* Xw1   = (const float*)d_in[2];
    const float* Xb1   = (const float*)d_in[3];
    const float* Xw2   = (const float*)d_in[4];
    const float* Xb2   = (const float*)d_in[5];
    const float* Xw3   = (const float*)d_in[6];
    const float* Xb3   = (const float*)d_in[7];
    const float* Xwo   = (const float*)d_in[8];
    const float* Xfreq = (const float*)d_in[9];
    const float* Yw1   = (const float*)d_in[10];
    const float* Yb1   = (const float*)d_in[11];
    const float* Yw2   = (const float*)d_in[12];
    const float* Yb2   = (const float*)d_in[13];
    const float* Yw3   = (const float*)d_in[14];
    const float* Yb3   = (const float*)d_in[15];
    const float* Ywo   = (const float*)d_in[16];
    const float* Yfreq = (const float*)d_in[17];
    const float* bias  = (const float*)d_in[18];
    float* out = (float*)d_out;

    // scratch: only 64 KB (hx, hy as bf16 [D][L])
    bf16_t* hx = (bf16_t*)d_ws;
    bf16_t* hy = (bf16_t*)((char*)d_ws + (size_t)DM * LL * sizeof(bf16_t));

    // 1) implicit filters (1/512 FFT normalization folded into hy)
    hyena_filters<<<64, 256, 0, stream>>>(z, Xw1, Xb1, Xw2, Xb2, Xw3, Xb3,
                                          Xwo, Xfreq, hx, 1.0f);
    hyena_filters<<<64, 256, 0, stream>>>(z, Yw1, Yb1, Yw2, Yb2, Yw3, Yb3,
                                          Ywo, Yfreq, hy, 1.0f / 512.0f);

    // 2) conv along last axis (hy) -> Y1 bf16 fragment layout (in d_out slots)
    conv_cols<<<NB * DM * 2, 256, 0, stream>>>(x, hy, out);

    // 3) conv along first axis (hx), async double-buffered, in-place + bias
    conv_rows<<<NB * DM, 512, 0, stream>>>(out, hx, x, bias);
}
